// MambaMIL_70987219468469
// MI455X (gfx1250) — compile-verified, matrix operand order checked by reading, not run on hardware
//
#include <hip/hip_runtime.h>
#include <hip/hip_bf16.h>
#include <math.h>

// ---------------------------------------------------------------------------
// MambaMIL forward for MI455X (gfx1250, wave32, WMMA).
// GEMMs: v_wmma_f32_16x16x32_bf16; fp32 -> bf16 tiles staged in LDS in
// fragment-friendly row-major [m|n][k] order so every WMMA operand is
// exactly two ds_load_b128 per lane. Block = 8 waves, tile 128x64,
// 4 WMMAs per wave per K-step.
// ---------------------------------------------------------------------------

typedef __attribute__((ext_vector_type(16))) __bf16 v16bf;
typedef __attribute__((ext_vector_type(8)))  float  v8f;

#define BM 128           // block tile M (8 waves x 16 rows)
#define BN 64            // block tile N (4 x 16-col subtiles per wave)
#define TK 32            // K step (bf16 WMMA depth)
#define LDP (TK + 8)     // padded LDS row stride in halves (80B, 16B-aligned)

union Pack8 { __bf16 h[8]; float4 f4; };     // 8 bf16 = 16 bytes
union FragBF { v16bf v; float4 f4[2]; };     // 16 bf16 = 32 bytes

// D[m,n] = act( sum_k A[m,k] * W[n,k] + bias[n] ) (+ residual[m,n])
// A: MxK (row stride lda), W: NxK row-major (stride K), D: MxN (row stride ldd)
// act: 0=none 1=relu 2=softplus 3=tanh
__global__ __launch_bounds__(256) void wmma_gemm_k(
    const float* __restrict__ A, int lda,
    const float* __restrict__ W, int K,
    const float* __restrict__ bias,
    const float* __restrict__ residual,
    float* __restrict__ D, int ldd,
    int act)
{
    __shared__ __bf16 aLds[BM][LDP];   // A tile, [m][k]
    __shared__ __bf16 bLds[BN][LDP];   // B tile as bT[n][k] (= W rows, no transpose)

    const int tid  = threadIdx.x;
    const int m0   = blockIdx.y * BM;
    const int n0   = blockIdx.x * BN;
    const int wave = tid >> 5;          // 0..7 -> 16-row stripe
    const int lane = tid & 31;
    const int g    = lane >> 4;         // half-wave group selects K-chunks
    const int ml   = lane & 15;
    const int g8   = g * 8;

    // staging assignments
    const int arow = tid >> 1;          // 0..127
    const int akq  = (tid & 1) * 16;    // 0 or 16
    const int bn   = tid >> 2;          // 0..63
    const int bkq  = (tid & 3) * 8;     // 0,8,16,24

    v8f acc[4];
    #pragma unroll
    for (int j = 0; j < 4; ++j)
        acc[j] = (v8f){0.f, 0.f, 0.f, 0.f, 0.f, 0.f, 0.f, 0.f};

    for (int kk = 0; kk < K; kk += TK) {
        // ---- stage A tile (128x32 fp32 -> bf16), 16 elems/thread ----
        {
            const float* ap = A + (size_t)(m0 + arow) * lda + kk + akq;
            #pragma unroll
            for (int q = 0; q < 2; ++q) {
                const float4 lo = *reinterpret_cast<const float4*>(ap + q * 8);
                const float4 hi = *reinterpret_cast<const float4*>(ap + q * 8 + 4);
                Pack8 p;
                p.h[0] = (__bf16)lo.x; p.h[1] = (__bf16)lo.y;
                p.h[2] = (__bf16)lo.z; p.h[3] = (__bf16)lo.w;
                p.h[4] = (__bf16)hi.x; p.h[5] = (__bf16)hi.y;
                p.h[6] = (__bf16)hi.z; p.h[7] = (__bf16)hi.w;
                *reinterpret_cast<float4*>(&aLds[arow][akq + q * 8]) = p.f4;
            }
            if (kk + TK < K)
                __builtin_prefetch(ap + TK, 0, 3);   // global_prefetch next A tile
        }
        // ---- stage B tile (64x32): straight copy of W rows (K-contiguous) ----
        {
            const float* wp = W + (size_t)(n0 + bn) * K + kk + bkq;
            const float4 lo = *reinterpret_cast<const float4*>(wp);
            const float4 hi = *reinterpret_cast<const float4*>(wp + 4);
            Pack8 p;
            p.h[0] = (__bf16)lo.x; p.h[1] = (__bf16)lo.y;
            p.h[2] = (__bf16)lo.z; p.h[3] = (__bf16)lo.w;
            p.h[4] = (__bf16)hi.x; p.h[5] = (__bf16)hi.y;
            p.h[6] = (__bf16)hi.z; p.h[7] = (__bf16)hi.w;
            *reinterpret_cast<float4*>(&bLds[bn][bkq]) = p.f4;
            if (kk + TK < K)
                __builtin_prefetch(wp + TK, 0, 3);
        }
        __syncthreads();

        // ---- fragments: lane l%16 = row, l/16 selects K-chunks {g8, 16+g8} ----
        FragBF fa;
        fa.f4[0] = *reinterpret_cast<const float4*>(&aLds[wave * 16 + ml][g8]);
        fa.f4[1] = *reinterpret_cast<const float4*>(&aLds[wave * 16 + ml][16 + g8]);

        #pragma unroll
        for (int j = 0; j < 4; ++j) {
            FragBF fb;
            fb.f4[0] = *reinterpret_cast<const float4*>(&bLds[j * 16 + ml][g8]);
            fb.f4[1] = *reinterpret_cast<const float4*>(&bLds[j * 16 + ml][16 + g8]);
            acc[j] = __builtin_amdgcn_wmma_f32_16x16x32_bf16(
                false, fa.v, false, fb.v, (short)0, acc[j], false, false);
        }
        __syncthreads();
    }

    // ---- epilogue: C/D layout: vgpr r -> row r + 8*g, col = lane%16 ----
    #pragma unroll
    for (int j = 0; j < 4; ++j) {
        const int col  = n0 + j * 16 + ml;
        const float bs = bias ? bias[col] : 0.f;
        #pragma unroll
        for (int r = 0; r < 8; ++r) {
            const int row = m0 + wave * 16 + g * 8 + r;
            float v = acc[j][r] + bs;
            if (act == 1)      v = fmaxf(v, 0.f);
            else if (act == 2) v = (v > 20.f) ? v : log1pf(__expf(v));
            else if (act == 3) v = tanhf(v);
            const size_t o = (size_t)row * ldd + col;
            if (residual) v += residual[o];
            D[o] = v;
        }
    }
}

// ---- layernorm over last dim (block per row) ----
__global__ void layernorm_k(const float* __restrict__ x,
                            const float* __restrict__ w,
                            const float* __restrict__ b,
                            float* __restrict__ out, int Dd)
{
    __shared__ float red[256];
    const int row = blockIdx.x;
    const int tid = threadIdx.x;
    const float* xr = x + (size_t)row * Dd;

    float s = 0.f;
    for (int i = tid; i < Dd; i += 256) s += xr[i];
    red[tid] = s; __syncthreads();
    for (int o = 128; o > 0; o >>= 1) {
        if (tid < o) red[tid] += red[tid + o];
        __syncthreads();
    }
    const float mean = red[0] / Dd;
    __syncthreads();

    float v = 0.f;
    for (int i = tid; i < Dd; i += 256) { float d = xr[i] - mean; v += d * d; }
    red[tid] = v; __syncthreads();
    for (int o = 128; o > 0; o >>= 1) {
        if (tid < o) red[tid] += red[tid + o];
        __syncthreads();
    }
    const float rstd = rsqrtf(red[0] / Dd + 1e-5f);

    for (int i = tid; i < Dd; i += 256)
        out[(size_t)row * Dd + i] = (xr[i] - mean) * rstd * w[i] + b[i];
}

// ---- causal depthwise conv (k=4) + bias + SiLU; xin = xz[:, :1024] ----
__global__ void conv_silu_k(const float* __restrict__ xz,
                            const float* __restrict__ cw,
                            const float* __restrict__ cb,
                            float* __restrict__ xs)
{
    const int idx = blockIdx.x * blockDim.x + threadIdx.x; // 4096*1024
    const int n = idx >> 10, d = idx & 1023;
    float acc = cb[d];
    #pragma unroll
    for (int j = 0; j < 4; ++j) {
        const int nn = n - 3 + j;
        if (nn >= 0) acc += xz[(size_t)nn * 2048 + d] * cw[d * 4 + j];
    }
    xs[idx] = acc / (1.f + __expf(-acc)); // silu
}

// ---- selective scan: thread per channel d, 16 states in regs.
//      delta_y holds softplus(dt@W+b) on entry, y*silu(z) on exit (in-place).
__global__ void scan_k(const float* __restrict__ xz,
                       const float* __restrict__ xs,
                       float* __restrict__ delta_y,
                       const float* __restrict__ dbc,
                       const float* __restrict__ Alog,
                       const float* __restrict__ Dv)
{
    const int d = blockIdx.x * blockDim.x + threadIdx.x; // 0..1023
    float Arow[16], st[16];
    #pragma unroll
    for (int s = 0; s < 16; ++s) {
        Arow[s] = -__expf(Alog[d * 16 + s]);
        st[s] = 0.f;
    }
    const float Dvd = Dv[d];
    for (int n = 0; n < 4096; ++n) {
        const float dt = delta_y[(size_t)n * 1024 + d];
        const float xv = xs[(size_t)n * 1024 + d];
        const float zv = xz[(size_t)n * 2048 + 1024 + d];
        const float* bc = dbc + (size_t)n * 64;
        float y = 0.f;
        #pragma unroll
        for (int s = 0; s < 16; ++s) {
            const float dA = __expf(dt * Arow[s]);
            st[s] = dA * st[s] + dt * bc[32 + s] * xv;
            y += st[s] * bc[48 + s];
        }
        y += Dvd * xv;
        const float sz = zv / (1.f + __expf(-zv)); // silu(z)
        delta_y[(size_t)n * 1024 + d] = y * sz;
    }
}

// ---- attention scores: s[n] = b2 + sum_k t[n,k]*w2[k] ----
__global__ void attn_score_k(const float* __restrict__ t,
                             const float* __restrict__ w2,
                             const float* __restrict__ b2,
                             float* __restrict__ sc)
{
    const int n = blockIdx.x * blockDim.x + threadIdx.x; // 4096
    float s = b2[0];
    #pragma unroll 4
    for (int k = 0; k < 128; ++k) s += t[(size_t)n * 128 + k] * w2[k];
    sc[n] = s;
}

// ---- single-block softmax over 4096 scores -> AM ----
__global__ void softmax_k(const float* __restrict__ sc, float* __restrict__ am)
{
    __shared__ float red[1024];
    const int tid = threadIdx.x;
    float mx = -1e30f;
    for (int i = tid; i < 4096; i += 1024) mx = fmaxf(mx, sc[i]);
    red[tid] = mx; __syncthreads();
    for (int o = 512; o > 0; o >>= 1) {
        if (tid < o) red[tid] = fmaxf(red[tid], red[tid + o]);
        __syncthreads();
    }
    mx = red[0]; __syncthreads();
    float sm = 0.f;
    for (int i = tid; i < 4096; i += 1024) sm += __expf(sc[i] - mx);
    red[tid] = sm; __syncthreads();
    for (int o = 512; o > 0; o >>= 1) {
        if (tid < o) red[tid] += red[tid + o];
        __syncthreads();
    }
    const float inv = 1.f / red[0];
    for (int i = tid; i < 4096; i += 1024) am[i] = __expf(sc[i] - mx) * inv;
}

// ---- pooled[d] = sum_n am[n] * hn[n,d] ----
__global__ void pool_k(const float* __restrict__ am,
                       const float* __restrict__ hn,
                       float* __restrict__ pooled)
{
    const int d = blockIdx.x * blockDim.x + threadIdx.x; // 512
    float s = 0.f;
    for (int n = 0; n < 4096; ++n) s += am[n] * hn[(size_t)n * 512 + d];
    pooled[d] = s;
}

// ---- classifier head: logits, sigmoid, softmax, argmax ----
__global__ void head_k(const float* __restrict__ pooled,
                       const float* __restrict__ cw,
                       const float* __restrict__ cb,
                       float* __restrict__ out)
{
    __shared__ float lg[2];
    const int tid = threadIdx.x;
    if (tid < 2) {
        float s = cb[tid];
        for (int k = 0; k < 512; ++k) s += pooled[k] * cw[tid * 512 + k];
        lg[tid] = s;
    }
    __syncthreads();
    if (tid == 0) {
        const float l0 = lg[0], l1 = lg[1];
        out[0] = 1.f / (1.f + __expf(-l0));      // sigmoid logits
        out[1] = 1.f / (1.f + __expf(-l1));
        const float m = fmaxf(l0, l1);
        const float e0 = __expf(l0 - m), e1 = __expf(l1 - m);
        const float inv = 1.f / (e0 + e1);
        out[4098] = e0 * inv;                    // Y_prob
        out[4099] = e1 * inv;
        out[4100] = (l1 > l0) ? 1.f : 0.f;       // Y_hat (argmax)
    }
}

extern "C" void kernel_launch(void* const* d_in, const int* in_sizes, int n_in,
                              void* d_out, int out_size, void* d_ws, size_t ws_size,
                              hipStream_t stream)
{
    const float* x     = (const float*)d_in[0];   // (1,4096,1024)
    const float* fc1_w = (const float*)d_in[1];   // (512,1024)
    const float* fc1_b = (const float*)d_in[2];   // (512)
    const float* ln_w  = (const float*)d_in[3];   // (2,512)
    const float* ln_b  = (const float*)d_in[4];
    const float* ipw   = (const float*)d_in[5];   // (2,2048,512)
    const float* cw    = (const float*)d_in[6];   // (2,1024,4)
    const float* cb    = (const float*)d_in[7];   // (2,1024)
    const float* xpw   = (const float*)d_in[8];   // (2,64,1024)
    const float* dtw   = (const float*)d_in[9];   // (2,1024,32)
    const float* dtb   = (const float*)d_in[10];  // (2,1024)
    const float* Alog  = (const float*)d_in[11];  // (2,1024,16)
    const float* Dv    = (const float*)d_in[12];  // (2,1024)
    const float* opw   = (const float*)d_in[13];  // (2,512,1024)
    const float* nw    = (const float*)d_in[14];  // (512)
    const float* nb    = (const float*)d_in[15];
    const float* aw1   = (const float*)d_in[16];  // (128,512)
    const float* ab1   = (const float*)d_in[17];
    const float* aw2   = (const float*)d_in[18];  // (1,128)
    const float* ab2   = (const float*)d_in[19];
    const float* clw   = (const float*)d_in[20];  // (2,512)
    const float* clb   = (const float*)d_in[21];

    float* ws  = (float*)d_ws;
    float* out = (float*)d_out;

    // workspace layout (all 16B-aligned offsets)
    float* hbuf   = ws;                             // 4096*512
    float* lnbuf  = hbuf   + (size_t)4096 * 512;    // 4096*512
    float* xzbuf  = lnbuf  + (size_t)4096 * 512;    // 4096*2048
    float* xsbuf  = xzbuf  + (size_t)4096 * 2048;   // 4096*1024
    float* dbcbuf = xsbuf  + (size_t)4096 * 1024;   // 4096*64
    float* dybuf  = dbcbuf + (size_t)4096 * 64;     // 4096*1024 (delta -> y)
    float* tbuf   = dybuf  + (size_t)4096 * 1024;   // 4096*128
    float* scbuf  = tbuf   + (size_t)4096 * 128;    // 4096
    float* pbuf   = scbuf  + 4096;                  // 512

    const dim3 blk(256);
    const int  gy = 4096 / BM;   // 32

    // h = relu(x @ fc1_w^T + fc1_b)          M=4096 N=512 K=1024
    wmma_gemm_k<<<dim3(512 / BN, gy), blk, 0, stream>>>(
        x, 1024, fc1_w, 1024, fc1_b, nullptr, hbuf, 512, 1);

    for (int l = 0; l < 2; ++l) {
        // ln = layernorm(h)
        layernorm_k<<<4096, 256, 0, stream>>>(hbuf, ln_w + l * 512, ln_b + l * 512, lnbuf, 512);
        // xz = ln @ in_proj_w^T               M=4096 N=2048 K=512
        wmma_gemm_k<<<dim3(2048 / BN, gy), blk, 0, stream>>>(
            lnbuf, 512, ipw + (size_t)l * 2048 * 512, 512, nullptr, nullptr, xzbuf, 2048, 0);
        // xs = silu(causal depthwise conv(xin) + cb)
        conv_silu_k<<<(4096 * 1024) / 256, 256, 0, stream>>>(
            xzbuf, cw + (size_t)l * 4096, cb + (size_t)l * 1024, xsbuf);
        // dbc = xs @ x_proj_w^T               M=4096 N=64 K=1024
        wmma_gemm_k<<<dim3(64 / BN, gy), blk, 0, stream>>>(
            xsbuf, 1024, xpw + (size_t)l * 64 * 1024, 1024, nullptr, nullptr, dbcbuf, 64, 0);
        // delta = softplus(dt @ dt_proj_w^T + dt_proj_b)  M=4096 N=1024 K=32 (A=dbc[:, :32], lda=64)
        wmma_gemm_k<<<dim3(1024 / BN, gy), blk, 0, stream>>>(
            dbcbuf, 64, dtw + (size_t)l * 1024 * 32, 32, dtb + (size_t)l * 1024, nullptr, dybuf, 1024, 2);
        // selective scan -> y (in-place over delta)
        scan_k<<<4, 256, 0, stream>>>(
            xzbuf, xsbuf, dybuf, dbcbuf, Alog + (size_t)l * 1024 * 16, Dv + (size_t)l * 1024);
        // h = h + y @ out_proj_w^T            M=4096 N=512 K=1024 (residual)
        wmma_gemm_k<<<dim3(512 / BN, gy), blk, 0, stream>>>(
            dybuf, 1024, opw + (size_t)l * 512 * 1024, 1024, nullptr, hbuf, hbuf, 512, 0);
    }

    // hn = layernorm(h)
    layernorm_k<<<4096, 256, 0, stream>>>(hbuf, nw, nb, lnbuf, 512);
    // t = tanh(hn @ attn_w1^T + b1)           M=4096 N=128 K=512
    wmma_gemm_k<<<dim3(128 / BN, gy), blk, 0, stream>>>(
        lnbuf, 512, aw1, 512, ab1, nullptr, tbuf, 128, 3);
    // scores
    attn_score_k<<<4096 / 256, 256, 0, stream>>>(tbuf, aw2, ab2, scbuf);
    // AM = softmax(scores) -> out[2..4097]
    softmax_k<<<1, 1024, 0, stream>>>(scbuf, out + 2);
    // pooled = AM^T @ hn
    pool_k<<<2, 256, 0, stream>>>(out + 2, lnbuf, pbuf);
    // head: logit/Y_prob/Y_hat
    head_k<<<1, 64, 0, stream>>>(pbuf, clw, clb, out);
}
